// GNNStack_16655883174257
// MI455X (gfx1250) — matrix operand order repaired and checked by reading, not compiled, not run on hardware
//
#include <hip/hip_runtime.h>
#include <hip/hip_bf16.h>

// ---------- constants ----------
#define N_NODES 50000
#define N_EDGES 800000
#define C_HID   128          // hidden width (= heads*out_ch for every GAT layer)
#define OUT_DIM 64

typedef __bf16 bf16;
typedef __attribute__((ext_vector_type(16))) __bf16 v16bf;
typedef __attribute__((ext_vector_type(8)))  __bf16 v8bf;
typedef __attribute__((ext_vector_type(8)))  float  v8f;

// ---------- helpers ----------
__device__ __forceinline__ unsigned f2key(float f) {
    unsigned u = __float_as_uint(f);
    return (u & 0x80000000u) ? ~u : (u | 0x80000000u);   // order-preserving key; real floats always > 0
}
__device__ __forceinline__ float key2f(unsigned k) {
    unsigned u = (k & 0x80000000u) ? (k & 0x7fffffffu) : ~k;
    return __uint_as_float(u);
}
__device__ __forceinline__ float leaky(float x) { return x > 0.f ? x : 0.2f * x; }

// ---------- f32 -> bf16 conversion ----------
__global__ void cvt_f32_bf16(const float* __restrict__ in, bf16* __restrict__ out, int n) {
    int i = blockIdx.x * blockDim.x + threadIdx.x;
    if (i < n) out[i] = (bf16)in[i];
}

// ---------- WMMA GEMM: out[M,N] = A[M,K] @ W[N,K]^T + bias ----------
// block = 128 threads (4 waves); wave w handles col tile (blockIdx.y*4 + w)
// M % 16 == 0, N % 64 == 0 or N==64, K % 32 == 0 required.
__global__ __launch_bounds__(128)
void gemm_bf16_wmma(const bf16* __restrict__ A, const bf16* __restrict__ W,
                    const float* __restrict__ bias, float* __restrict__ out,
                    int K, int N, int do_relu) {
    const int lane   = threadIdx.x & 31;
    const int wave   = threadIdx.x >> 5;
    const int rowBase = blockIdx.x * 16;
    const int colBase = (blockIdx.y * 4 + wave) * 16;
    const int half = lane >> 4;     // 0: lanes 0-15, 1: lanes 16-31
    const int l15  = lane & 15;

    const bf16* arow = A + (size_t)(rowBase + l15) * K;
    const bf16* brow = W + (size_t)(colBase + l15) * K;

    v8f acc = {};
    for (int k0 = 0; k0 < K; k0 += 32) {
        // A 16x32 tile: two 16B chunks per lane (ISA 7.12.2 16-bit A layout)
        const v8bf* pa = reinterpret_cast<const v8bf*>(arow + k0 + 8 * half);
        v8bf alo = pa[0];          // K = k0 + 8*half + 0..7   -> VGPR 0..3
        v8bf ahi = pa[2];          // K = k0 + 8*half + 16..23 -> VGPR 4..7
        v16bf a;
#pragma unroll
        for (int i = 0; i < 8; ++i) { a[i] = alo[i]; a[i + 8] = ahi[i]; }
        // B 32x16 tile: B[k][n] = W[n][k]; lane holds column l15, same K chunks as A
        const v8bf* pb = reinterpret_cast<const v8bf*>(brow + k0 + 8 * half);
        v8bf blo = pb[0];
        v8bf bhi = pb[2];
        v16bf b;
#pragma unroll
        for (int i = 0; i < 8; ++i) { b[i] = blo[i]; b[i + 8] = bhi[i]; }
        acc = __builtin_amdgcn_wmma_f32_16x16x32_bf16(
                  false, a, false, b, (short)0, acc, false, false);
    }

    const int col = colBase + l15;
    const float bv = bias ? bias[col] : 0.f;
#pragma unroll
    for (int r = 0; r < 8; ++r) {
        int row = rowBase + r + 8 * half;
        float v = acc[r] + bv;
        if (do_relu) v = v > 0.f ? v : 0.f;
        out[(size_t)row * N + col] = v;
    }
}

// ---------- attention projections: a_i / a_j per (node, head) ----------
__global__ void att_scores(const float* __restrict__ h, const float* __restrict__ att,
                           float* __restrict__ ai, float* __restrict__ aj,
                           int n, int H, int OC) {
    int id = blockIdx.x * blockDim.x + threadIdx.x;     // id = node*H + head
    if (id >= n * H) return;
    int head = id % H;
    const float* hp  = h + (size_t)id * OC;             // == node*128 + head*OC (H*OC==128)
    const float* ati = att + head * 2 * OC;
    const float* atj = ati + OC;
    float si = 0.f, sj = 0.f;
    for (int c = 0; c < OC; ++c) { float v = hp[c]; si += v * ati[c]; sj += v * atj[c]; }
    ai[id] = si; aj[id] = sj;
}

// ---------- edge pass 1: segment max of leaky(a_i[dst]+a_j[src]) ----------
__global__ void edge_max(const int* __restrict__ src, const int* __restrict__ dst,
                         const float* __restrict__ ai, const float* __restrict__ aj,
                         unsigned* __restrict__ amaxk, int E, int H) {
    int id = blockIdx.x * blockDim.x + threadIdx.x;
    if (id >= E * H) return;
    int e = id / H, hh = id - e * H;
    int s = src[e], d = dst[e];
    float al = leaky(ai[d * H + hh] + aj[s * H + hh]);
    atomicMax(&amaxk[d * H + hh], f2key(al));
}

// ---------- edge pass 2: segment sum of exp(alpha - max) ----------
__global__ void edge_denom(const int* __restrict__ src, const int* __restrict__ dst,
                           const float* __restrict__ ai, const float* __restrict__ aj,
                           const unsigned* __restrict__ amaxk, float* __restrict__ denom,
                           int E, int H) {
    int id = blockIdx.x * blockDim.x + threadIdx.x;
    if (id >= E * H) return;
    int e = id / H, hh = id - e * H;
    int s = src[e], d = dst[e];
    float al = leaky(ai[d * H + hh] + aj[s * H + hh]);
    float ex = __expf(al - key2f(amaxk[d * H + hh]));
    atomicAdd(&denom[d * H + hh], ex);
}

// ---------- edge pass 3: agg[dst] += h[src] * alpha  (32 lanes per edge, 4 ch each) ----------
__global__ void edge_aggregate(const int* __restrict__ src, const int* __restrict__ dst,
                               const float* __restrict__ ai, const float* __restrict__ aj,
                               const unsigned* __restrict__ amaxk, const float* __restrict__ denom,
                               const float* __restrict__ h, float* __restrict__ agg,
                               int E, int H, int logOC) {
    int id   = blockIdx.x * blockDim.x + threadIdx.x;
    int e    = id >> 5;
    int lane = id & 31;
    if (e >= E) return;
    int s = src[e], d = dst[e];
    const float* hs = h + (size_t)s * C_HID;
    float* ag = agg + (size_t)d * C_HID;
#pragma unroll
    for (int j = 0; j < 4; ++j) {
        int c  = lane + 32 * j;
        int hh = c >> logOC;
        float al   = leaky(ai[d * H + hh] + aj[s * H + hh]);
        float ex   = __expf(al - key2f(amaxk[d * H + hh]));
        float coef = ex / (denom[d * H + hh] + 1e-16f);
        atomicAdd(&ag[c], hs[c] * coef);
    }
}

// ---------- epilogue: h = relu(agg + bias) ----------
__global__ void bias_relu(const float* __restrict__ agg, const float* __restrict__ bias,
                          float* __restrict__ out, int n) {
    int i = blockIdx.x * blockDim.x + threadIdx.x;
    if (i >= n) return;
    float v = agg[i] + bias[i & (C_HID - 1)];
    out[i] = v > 0.f ? v : 0.f;
}

// ---------- log_softmax over 64 cols, one wave per row ----------
__global__ void log_softmax64(float* __restrict__ io, int nrows) {
    int gid  = blockIdx.x * blockDim.x + threadIdx.x;
    int row  = gid >> 5;
    int lane = gid & 31;
    if (row >= nrows) return;
    float* p = io + (size_t)row * OUT_DIM;
    float v0 = p[lane], v1 = p[lane + 32];
    float m = fmaxf(v0, v1);
#pragma unroll
    for (int o = 16; o > 0; o >>= 1) m = fmaxf(m, __shfl_xor(m, o, 32));
    float ssum = __expf(v0 - m) + __expf(v1 - m);
#pragma unroll
    for (int o = 16; o > 0; o >>= 1) ssum += __shfl_xor(ssum, o, 32);
    float lse = m + __logf(ssum);
    p[lane] = v0 - lse;
    p[lane + 32] = v1 - lse;
}

// ---------- host orchestration ----------
extern "C" void kernel_launch(void* const* d_in, const int* in_sizes, int n_in,
                              void* d_out, int out_size, void* d_ws, size_t ws_size,
                              hipStream_t stream) {
    (void)in_sizes; (void)n_in; (void)out_size; (void)ws_size;
    const int N = N_NODES, E = N_EDGES;

    const float* x   = (const float*)d_in[0];
    const int*   ei  = (const int*)d_in[1];
    const int*   src = ei;
    const int*   dst = ei + E;
    const float* Wl[3]   = { (const float*)d_in[2],  (const float*)d_in[6],  (const float*)d_in[10] };
    const float* bl[3]   = { (const float*)d_in[3],  (const float*)d_in[7],  (const float*)d_in[11] };
    const float* attl[3] = { (const float*)d_in[4],  (const float*)d_in[8],  (const float*)d_in[12] };
    const float* bias[3] = { (const float*)d_in[5],  (const float*)d_in[9],  (const float*)d_in[13] };
    const float* Wp1 = (const float*)d_in[14];
    const float* bp1 = (const float*)d_in[15];
    const float* Wp2 = (const float*)d_in[16];
    const float* bp2 = (const float*)d_in[17];

    // workspace carving (256B aligned)
    char* wp = (char*)d_ws;
    auto carve = [&](size_t bytes) -> void* {
        void* r = (void*)wp;
        wp += (bytes + 255) & ~(size_t)255;
        return r;
    };
    float*    h      = (float*)carve((size_t)N * C_HID * 4);   // activations / GEMM out
    float*    agg    = (float*)carve((size_t)N * C_HID * 4);   // aggregation + t1 reuse
    bf16*     abf    = (bf16*)carve((size_t)N * C_HID * 2);    // bf16 activation staging
    float*    ai     = (float*)carve((size_t)N * 8 * 4);
    float*    aj     = (float*)carve((size_t)N * 8 * 4);
    unsigned* amaxk  = (unsigned*)carve((size_t)N * 8 * 4);
    float*    denom  = (float*)carve((size_t)N * 8 * 4);
    bf16*     wbf[5];
    wbf[0] = (bf16*)carve(128 * 128 * 2);
    wbf[1] = (bf16*)carve(128 * 128 * 2);
    wbf[2] = (bf16*)carve(128 * 128 * 2);
    wbf[3] = (bf16*)carve(128 * 128 * 2);
    wbf[4] = (bf16*)carve(OUT_DIM * 128 * 2);

    const int TPB = 256;
    // weight conversions (every call; deterministic)
    cvt_f32_bf16<<<(128 * 128 + TPB - 1) / TPB, TPB, 0, stream>>>(Wl[0], wbf[0], 128 * 128);
    cvt_f32_bf16<<<(128 * 128 + TPB - 1) / TPB, TPB, 0, stream>>>(Wl[1], wbf[1], 128 * 128);
    cvt_f32_bf16<<<(128 * 128 + TPB - 1) / TPB, TPB, 0, stream>>>(Wl[2], wbf[2], 128 * 128);
    cvt_f32_bf16<<<(128 * 128 + TPB - 1) / TPB, TPB, 0, stream>>>(Wp1,   wbf[3], 128 * 128);
    cvt_f32_bf16<<<(OUT_DIM * 128 + TPB - 1) / TPB, TPB, 0, stream>>>(Wp2, wbf[4], OUT_DIM * 128);

    const int nElem = N * C_HID;
    const dim3 g128(N / 16, 128 / 64);     // N=128 output cols
    const dim3 g64 (N / 16, 64 / 64);      // N=64 output cols

    // layer-0 input GEMM
    cvt_f32_bf16<<<(nElem + TPB - 1) / TPB, TPB, 0, stream>>>(x, abf, nElem);
    gemm_bf16_wmma<<<g128, 128, 0, stream>>>(abf, wbf[0], bl[0], h, 128, 128, 0);

    const int Hh[3]    = { 8, 1, 1 };
    const int OC[3]    = { 16, 128, 128 };
    const int logOC[3] = { 4, 7, 7 };

    for (int l = 0; l < 3; ++l) {
        const int H = Hh[l];
        hipMemsetAsync(amaxk, 0, (size_t)N * H * 4, stream);
        hipMemsetAsync(denom, 0, (size_t)N * H * 4, stream);
        hipMemsetAsync(agg,   0, (size_t)nElem * 4, stream);

        att_scores<<<(N * H + TPB - 1) / TPB, TPB, 0, stream>>>(h, attl[l], ai, aj, N, H, OC[l]);
        edge_max  <<<(E * H + TPB - 1) / TPB, TPB, 0, stream>>>(src, dst, ai, aj, amaxk, E, H);
        edge_denom<<<(E * H + TPB - 1) / TPB, TPB, 0, stream>>>(src, dst, ai, aj, amaxk, denom, E, H);
        edge_aggregate<<<((size_t)E * 32 + TPB - 1) / TPB, TPB, 0, stream>>>(
            src, dst, ai, aj, amaxk, denom, h, agg, E, H, logOC[l]);
        bias_relu<<<(nElem + TPB - 1) / TPB, TPB, 0, stream>>>(agg, bias[l], h, nElem);

        if (l < 2) {
            cvt_f32_bf16<<<(nElem + TPB - 1) / TPB, TPB, 0, stream>>>(h, abf, nElem);
            gemm_bf16_wmma<<<g128, 128, 0, stream>>>(abf, wbf[l + 1], bl[l + 1], h, 128, 128, 0);
        }
    }

    // MLP head: t1 = h @ Wp1^T + bp1 (no relu), logits = t1 @ Wp2^T + bp2
    cvt_f32_bf16<<<(nElem + TPB - 1) / TPB, TPB, 0, stream>>>(h, abf, nElem);
    gemm_bf16_wmma<<<g128, 128, 0, stream>>>(abf, wbf[3], bp1, agg, 128, 128, 0);
    cvt_f32_bf16<<<(nElem + TPB - 1) / TPB, TPB, 0, stream>>>(agg, abf, nElem);
    gemm_bf16_wmma<<<g64, 128, 0, stream>>>(abf, wbf[4], bp2, (float*)d_out, 128, OUT_DIM, 0);

    // row-wise log_softmax over 64 logits (in-place on d_out)
    log_softmax64<<<((size_t)N * 32 + TPB - 1) / TPB, TPB, 0, stream>>>((float*)d_out, N);
}
